// CrossAttention_tau_90958817394713
// MI455X (gfx1250) — matrix operand order repaired and checked by reading, not compiled, not run on hardware
//
#include <hip/hip_runtime.h>
#include <hip/hip_bf16.h>

// ---------------------------------------------------------------------------
// CDNA5 (gfx1250) cross-attention with learned temperature.
// B=4, N=2048, C=768, H=12, D=64.  All matmuls via v_wmma_f32_16x16x32_bf16.
// Attention K/V tiles staged to LDS with global_load_async_to_lds_b128
// (ASYNCcnt), double-buffered across the 32-key block loop.
// ---------------------------------------------------------------------------

typedef __attribute__((ext_vector_type(16))) __bf16 v16bf;
typedef __attribute__((ext_vector_type(8)))  float  v8f;

constexpr int   Bv    = 4;
constexpr int   Nv    = 2048;
constexpr int   Cv    = 768;
constexpr int   Hv    = 12;
constexpr int   Dv    = 64;
constexpr int   BN    = Bv * Nv;                 // 8192 rows
constexpr float SCALE = 0.125f;                  // D^-0.5
constexpr float LOG2E = 1.44269504088896340736f;

// --- small POD helpers (guaranteed trivially-copyable for bit_cast) --------
struct alignas(16) U128 { unsigned int w[4]; };
struct Frag16 { U128 lo, hi; };                   // 32 bytes = one v16bf operand
struct alignas(16) F4 { float x, y, z, w; };
struct U64p { unsigned int lo, hi; };

static __device__ __forceinline__ unsigned short f2bf(float x) {
  unsigned int u = __builtin_bit_cast(unsigned int, x);
  u += 0x7FFFu + ((u >> 16) & 1u);                // round-to-nearest-even
  return (unsigned short)(u >> 16);
}

static __device__ __forceinline__ v8f zero8() {
  v8f v;
#pragma unroll
  for (int i = 0; i < 8; ++i) v[i] = 0.0f;
  return v;
}

static __device__ __forceinline__ v8f wmma_bf16(const Frag16 &a, const Frag16 &b, v8f c) {
  return __builtin_amdgcn_wmma_f32_16x16x32_bf16(
      false, __builtin_bit_cast(v16bf, a),
      false, __builtin_bit_cast(v16bf, b),
      (short)0, c, false, false);
}

// A-matrix fragment: 16(M) x 32(K) bf16 from row-major src (ld elements).
// ISA layout: lanes 0-15 hold K=[0..8)+[16..24), lanes 16-31 K=[8..16)+[24..32).
static __device__ __forceinline__ Frag16 load_a(const unsigned short *src,
                                                int row0, int k0, int ld, int lane) {
  int m  = lane & 15;
  int kg = (lane >> 4) << 3;                       // 0 or 8
  const unsigned short *p = src + (size_t)(row0 + m) * ld + k0 + kg;
  Frag16 f;
  f.lo = *reinterpret_cast<const U128 *>(p);       // K = k0+kg    .. +8
  f.hi = *reinterpret_cast<const U128 *>(p + 16);  // K = k0+16+kg .. +8
  return f;
}

// B-matrix fragment: 32(K) x 16(N) bf16.  Source is the row-major transposed
// operand W[n,k] (ld elements): lane holds column n with a contiguous K run.
// ISA layout: lanes 0-15 hold K=[0..8)+[16..24), lanes 16-31 K=[8..16)+[24..32).
static __device__ __forceinline__ Frag16 load_b(const unsigned short *src,
                                                int n0, int k0, int ld, int lane) {
  int n  = n0 + (lane & 15);
  int kb = (lane >> 4) << 3;                       // 0 or 8
  const unsigned short *p = src + (size_t)n * ld + k0 + kb;
  Frag16 f;
  f.lo = *reinterpret_cast<const U128 *>(p);
  f.hi = *reinterpret_cast<const U128 *>(p + 16);
  return f;
}

// --- CDNA5 async copy: global -> LDS, 16B per lane, tracked by ASYNCcnt ----
static __device__ __forceinline__ unsigned int lds_off32(const void *p) {
  // generic LDS address: low 32 bits are the wave-relative LDS byte offset
  return (unsigned int)(unsigned long long)p;
}
static __device__ __forceinline__ void async_b128(unsigned int lds_off,
                                                  const unsigned short *g) {
  asm volatile("global_load_async_to_lds_b128 %0, %1, off"
               :: "v"(lds_off), "v"(g) : "memory");
}

// ---------------------------------------------------------------------------
// fp32 -> bf16 conversion (vectorized x4)
// ---------------------------------------------------------------------------
__global__ void cvt_f32_bf16_kernel(const float *__restrict__ src,
                                    unsigned short *__restrict__ dst, int n4) {
  int i = blockIdx.x * blockDim.x + threadIdx.x;
  if (i >= n4) return;
  F4 v = reinterpret_cast<const F4 *>(src)[i];
  U64p o;
  o.lo = (unsigned int)f2bf(v.x) | ((unsigned int)f2bf(v.y) << 16);
  o.hi = (unsigned int)f2bf(v.z) | ((unsigned int)f2bf(v.w) << 16);
  reinterpret_cast<U64p *>(dst)[i] = o;
}

// ---------------------------------------------------------------------------
// Projection GEMM: dst = A[BN,Cv] @ W[Cv,Cv]^T  (all bf16, f32 accumulate)
// mode 0: write [B,H,N,D]   mode 1: write transposed [B,H,D,N] (for V)
// Block = 128 threads = 4 waves; each wave owns one 16x16 output tile.
// ---------------------------------------------------------------------------
__global__ __launch_bounds__(128) void proj_gemm_kernel(
    const unsigned short *__restrict__ A,
    const unsigned short *__restrict__ W,
    unsigned short *__restrict__ dst, int mode) {
  const int lane = threadIdx.x & 31;
  const int wave = threadIdx.x >> 5;
  const int row0 = blockIdx.x * 16;
  const int col0 = blockIdx.y * 64 + wave * 16;

  v8f acc = zero8();
  for (int k0 = 0; k0 < Cv; k0 += 32) {
    Frag16 fa = load_a(A, row0, k0, Cv, lane);
    Frag16 fb = load_b(W, col0, k0, Cv, lane);
    acc = wmma_bf16(fa, fb, acc);
  }

  const int rowoff = (lane >> 4) << 3;
  const int col    = lane & 15;
#pragma unroll
  for (int r = 0; r < 8; ++r) {
    int grow = row0 + r + rowoff;                  // global row in [0, BN)
    int gcol = col0 + col;                         // out channel h*D+d
    int b = grow >> 11, n = grow & (Nv - 1);
    int h = gcol >> 6,  d = gcol & (Dv - 1);
    unsigned short v = f2bf(acc[r]);
    if (mode == 0)
      dst[(size_t)((b * Hv + h) * Nv + n) * Dv + d] = v;
    else
      dst[(size_t)((b * Hv + h) * Dv + d) * Nv + n] = v;
  }
}

// ---------------------------------------------------------------------------
// Flash attention: per-(b,h), one wave = 16 query rows, 32-key blocks.
// Q,K: [B,H,N,D] bf16.  Vt: [B,H,D,N] bf16 (pre-transposed -> contiguous
// B-fragment loads for P@V).  AO: [B,N,C] bf16.
// K/V tiles are shared by all 4 waves -> staged once per block into LDS via
// async-to-LDS copies, double-buffered against the WMMA+softmax pipeline.
// ---------------------------------------------------------------------------
constexpr int KT = 32;                 // keys per block iteration
constexpr int NT = Nv / KT;            // 64 iterations

// issue one tile's async copies: K tile 32x64 (contiguous 4KB) + V tile 64x32
// (64 strided rows of 64B).  4 wave-instructions per wave per tile.
static __device__ __forceinline__ void stage_kv(const unsigned short *Kb,
                                                const unsigned short *Vb,
                                                unsigned short *ksh,
                                                unsigned short *vsh,
                                                int kk, int tid) {
#pragma unroll
  for (int r = 0; r < 2; ++r) {
    int c = tid + r * 128;                         // 16B chunk id, 0..255
    async_b128(lds_off32(ksh + c * 8), Kb + (size_t)kk * Dv + c * 8);
  }
#pragma unroll
  for (int r = 0; r < 2; ++r) {
    int c = tid + r * 128;
    int row = c >> 2, coff = (c & 3) * 8;          // d-row, 8-elem chunk
    async_b128(lds_off32(vsh + row * KT + coff),
               Vb + (size_t)row * Nv + kk + coff);
  }
}

__global__ __launch_bounds__(128) void attn_fa_kernel(
    const unsigned short *__restrict__ Q,
    const unsigned short *__restrict__ Km,
    const unsigned short *__restrict__ Vt,
    const float *__restrict__ tau_p,
    unsigned short *__restrict__ AO) {
  const int tid   = threadIdx.x;
  const int lane  = tid & 31;
  const int wave  = tid >> 5;
  const int bh    = blockIdx.x;                    // b*H + h
  const int b     = bh / Hv, h = bh % Hv;
  const int qrow0 = blockIdx.y * 64 + wave * 16;

  const unsigned short *Qb = Q  + (size_t)bh * Nv * Dv;
  const unsigned short *Kb = Km + (size_t)bh * Nv * Dv;
  const unsigned short *Vb = Vt + (size_t)bh * Dv * Nv;

  const float tau = log1pf(expf(tau_p[0])) + 1e-6f;   // softplus(tau)+eps
  const float sc2 = SCALE / tau * LOG2E;              // folded into exp2 domain

  const Frag16 a0 = load_a(Qb, qrow0, 0,  Dv, lane);
  const Frag16 a1 = load_a(Qb, qrow0, 32, Dv, lane);

  float mrow[8], lrow[8];
#pragma unroll
  for (int r = 0; r < 8; ++r) { mrow[r] = -3.0e38f; lrow[r] = 0.0f; }
  v8f o[4];
#pragma unroll
  for (int j = 0; j < 4; ++j) o[j] = zero8();

  __shared__ __align__(16) unsigned short Ksh[2][KT * Dv];   // 2 x 4KB
  __shared__ __align__(16) unsigned short Vsh[2][Dv * KT];   // 2 x 4KB
  __shared__ __align__(16) unsigned short plds[4][16 * 32];  // P tile per wave
  unsigned short *pw = plds[wave];

  const int rowoff = (lane >> 4) << 3;
  const int col    = lane & 15;

  stage_kv(Kb, Vb, Ksh[0], Vsh[0], 0, tid);        // prologue: tile 0

  for (int it = 0; it < NT; ++it) {
    const unsigned short *ksh = Ksh[it & 1];
    const unsigned short *vsh = Vsh[it & 1];
    if (it + 1 < NT) {
      stage_kv(Kb, Vb, Ksh[(it + 1) & 1], Vsh[(it + 1) & 1], (it + 1) * KT, tid);
      // 4 just-issued ops outstanding -> current tile's 4 are complete
      asm volatile("s_wait_asynccnt 0x4" ::: "memory");
    } else {
      asm volatile("s_wait_asynccnt 0x0" ::: "memory");
    }
    __syncthreads();                               // cross-wave visibility

    // S tile: 16 queries x 32 keys, contract over d=64 (two k=32 WMMA steps)
    v8f s0 = zero8(), s1 = zero8();
    s0 = wmma_bf16(a0, load_b(ksh, 0,  0,  Dv, lane), s0);
    s0 = wmma_bf16(a1, load_b(ksh, 0,  32, Dv, lane), s0);
    s1 = wmma_bf16(a0, load_b(ksh, 16, 0,  Dv, lane), s1);
    s1 = wmma_bf16(a1, load_b(ksh, 16, 32, Dv, lane), s1);

    float corr[8];
#pragma unroll
    for (int r = 0; r < 8; ++r) {
      float x0 = s0[r] * sc2, x1 = s1[r] * sc2;
      float mx = fmaxf(x0, x1);                     // row max over 32 keys
      mx = fmaxf(mx, __shfl_xor(mx, 1, 16));
      mx = fmaxf(mx, __shfl_xor(mx, 2, 16));
      mx = fmaxf(mx, __shfl_xor(mx, 4, 16));
      mx = fmaxf(mx, __shfl_xor(mx, 8, 16));
      float mnew = fmaxf(mrow[r], mx);
      float p0 = exp2f(x0 - mnew);
      float p1 = exp2f(x1 - mnew);
      float ps = p0 + p1;                           // row sum
      ps += __shfl_xor(ps, 1, 16);
      ps += __shfl_xor(ps, 2, 16);
      ps += __shfl_xor(ps, 4, 16);
      ps += __shfl_xor(ps, 8, 16);
      float c  = exp2f(mrow[r] - mnew);
      mrow[r]  = mnew;
      lrow[r]  = lrow[r] * c + ps;
      corr[r]  = c;
      // C-frag (row = r+rowoff, cols col / col+16) -> LDS P tile (bf16)
      pw[(r + rowoff) * 32 + col]      = f2bf(p0);
      pw[(r + rowoff) * 32 + col + 16] = f2bf(p1);
    }
#pragma unroll
    for (int j = 0; j < 4; ++j)
#pragma unroll
      for (int r = 0; r < 8; ++r) o[j][r] *= corr[r];

    // cross-lane LDS RAW inside the wave: drain ds_store_b16 before ds_load
    asm volatile("s_wait_dscnt 0" ::: "memory");
    const Frag16 pa = load_a(pw, 0, 0, 32, lane);   // P as A-fragment 16x32

#pragma unroll
    for (int j = 0; j < 4; ++j)                     // O += P @ V (d tiles of 16)
      o[j] = wmma_bf16(pa, load_b(vsh, j * 16, 0, KT, lane), o[j]);

    __syncthreads();   // all waves done with ksh/vsh before next overwrite
  }

  float inv[8];
#pragma unroll
  for (int r = 0; r < 8; ++r) inv[r] = 1.0f / lrow[r];
#pragma unroll
  for (int j = 0; j < 4; ++j) {
#pragma unroll
    for (int r = 0; r < 8; ++r) {
      int grow = qrow0 + r + rowoff;
      int gcol = h * Dv + j * 16 + col;
      AO[(size_t)(b * Nv + grow) * Cv + gcol] = f2bf(o[j][r] * inv[r]);
    }
  }
}

// ---------------------------------------------------------------------------
// Output projection: out[BN,Cv] = AO @ Wproj^T + bproj   (fp32 result)
// ---------------------------------------------------------------------------
__global__ __launch_bounds__(128) void out_gemm_kernel(
    const unsigned short *__restrict__ A,
    const unsigned short *__restrict__ W,
    const float *__restrict__ bias,
    float *__restrict__ out) {
  const int lane = threadIdx.x & 31;
  const int wave = threadIdx.x >> 5;
  const int row0 = blockIdx.x * 16;
  const int col0 = blockIdx.y * 64 + wave * 16;

  v8f acc = zero8();
  for (int k0 = 0; k0 < Cv; k0 += 32) {
    Frag16 fa = load_a(A, row0, k0, Cv, lane);
    Frag16 fb = load_b(W, col0, k0, Cv, lane);
    acc = wmma_bf16(fa, fb, acc);
  }

  const int rowoff = (lane >> 4) << 3;
  const int col    = lane & 15;
  const int gcol   = col0 + col;
  const float bia  = bias[gcol];
#pragma unroll
  for (int r = 0; r < 8; ++r) {
    int grow = row0 + r + rowoff;
    out[(size_t)grow * Cv + gcol] = acc[r] + bia;
  }
}

// ---------------------------------------------------------------------------
extern "C" void kernel_launch(void *const *d_in, const int *in_sizes, int n_in,
                              void *d_out, int out_size, void *d_ws, size_t ws_size,
                              hipStream_t stream) {
  const float *x     = (const float *)d_in[0];
  const float *y     = (const float *)d_in[1];
  const float *Wq    = (const float *)d_in[2];
  const float *Wkv   = (const float *)d_in[3];
  const float *tau   = (const float *)d_in[4];
  const float *Wproj = (const float *)d_in[5];
  const float *bproj = (const float *)d_in[6];
  float *out = (float *)d_out;

  // workspace carve-up (bf16 elements); total ~80.2 MB
  unsigned short *ws = (unsigned short *)d_ws;
  size_t off = 0;
  unsigned short *xbf   = ws + off; off += (size_t)BN * Cv;
  unsigned short *ybf   = ws + off; off += (size_t)BN * Cv;
  unsigned short *wqbf  = ws + off; off += (size_t)Cv * Cv;
  unsigned short *wkvbf = ws + off; off += (size_t)2 * Cv * Cv;
  unsigned short *wpbf  = ws + off; off += (size_t)Cv * Cv;
  unsigned short *Qb    = ws + off; off += (size_t)BN * Cv;   // [B,H,N,D]
  unsigned short *Kb    = ws + off; off += (size_t)BN * Cv;   // [B,H,N,D]
  unsigned short *Vtb   = ws + off; off += (size_t)BN * Cv;   // [B,H,D,N]
  unsigned short *AO    = ws + off; off += (size_t)BN * Cv;   // [B,N,C]

  auto cvt = [&](const float *s, unsigned short *d, size_t n) {
    int n4 = (int)(n / 4);
    cvt_f32_bf16_kernel<<<(n4 + 255) / 256, 256, 0, stream>>>(s, d, n4);
  };
  cvt(x,     xbf,   (size_t)BN * Cv);
  cvt(y,     ybf,   (size_t)BN * Cv);
  cvt(Wq,    wqbf,  (size_t)Cv * Cv);
  cvt(Wkv,   wkvbf, (size_t)2 * Cv * Cv);
  cvt(Wproj, wpbf,  (size_t)Cv * Cv);

  dim3 pg(BN / 16, Cv / 64);                       // 512 x 12 blocks
  proj_gemm_kernel<<<pg, 128, 0, stream>>>(xbf, wqbf, Qb, 0);
  proj_gemm_kernel<<<pg, 128, 0, stream>>>(ybf, wkvbf, Kb, 0);
  proj_gemm_kernel<<<pg, 128, 0, stream>>>(ybf, wkvbf + (size_t)Cv * Cv, Vtb, 1);

  dim3 ag(Bv * Hv, Nv / 64);                       // 48 x 32 blocks
  attn_fa_kernel<<<ag, 128, 0, stream>>>(Qb, Kb, Vtb, tau, AO);

  out_gemm_kernel<<<pg, 128, 0, stream>>>(AO, wpbf, bproj, out);
}
